// TreeFilter2D_13623636263194
// MI455X (gfx1250) — matrix operand order repaired and checked by reading, not compiled, not run on hardware
//
#include <hip/hip_runtime.h>
#include <hip/hip_bf16.h>

// ---- problem constants (fixed by the reference) ----
#define B_    4
#define C_    256      // feature channels (norm channel stored at index C_)
#define CE_   64       // embedding channels
#define V_    16384    // H*W
#define NL_   7        // tree depth
#define SIGMA_ 0.02f
#define CS_   260      // padded row stride of T (257 used, rows 16B-aligned)

typedef __attribute__((ext_vector_type(16))) _Float16 v16h;
typedef __attribute__((ext_vector_type(8)))  float    v8f;

// ---------------------------------------------------------------------------
// Edge weights: ew[b,v] = exp(-||em[:,si[v]] - em[:,si[parent[v]]]||^2 / sigma)
// One wave handles 16 BFS nodes. The 16x64 f16 diff tile is staged in LDS,
// then D = diff * diff^T via two v_wmma_f32_16x16x32_f16 (K = 64 total);
// diag(D) gives the 16 squared distances.
// ---------------------------------------------------------------------------
__global__ void tf_ew_wmma_kernel(const float* __restrict__ embed,
                                  const int*   __restrict__ si,
                                  const int*   __restrict__ sp,
                                  float*       __restrict__ ew)
{
    __shared__ _Float16 sdiff[4][16][72]; // [wave][node][ce], padded
    __shared__ float    sD[4][16][17];    // [wave][M][N], padded

    const int wv   = threadIdx.x >> 5;
    const int lane = threadIdx.x & 31;
    const int g    = blockIdx.x * 64 + wv * 16;   // first node of this wave
    const int b    = g / V_;
    const int v0   = g - b * V_;

    // Load diff tile: lane pair (2i,2i+1) covers node i, 32 channels each.
    {
        const int i    = lane >> 1;
        const int half = lane & 1;
        const int v    = v0 + i;
        const int pixs = si[b * V_ + v];
        const int par  = sp[b * V_ + v];
        const int pixt = si[b * V_ + par];
        const float* eb = embed + (size_t)b * CE_ * V_;
        #pragma unroll
        for (int k = 0; k < 32; ++k) {
            const int c  = half * 32 + k;
            const float d = eb[(size_t)c * V_ + pixs] - eb[(size_t)c * V_ + pixt];
            sdiff[wv][i][c] = (_Float16)d;
        }
    }
    __syncthreads();

    // Build A (16x32) and B = A^T (32x16) fragments in contiguous k order, same for A and B.
    const int m  = lane & 15;
    const int hi = lane >> 4;
    v8f acc = {};
    #pragma unroll
    for (int chunk = 0; chunk < 2; ++chunk) {
        const int kofs = chunk * 32;
        v16h a, bm;
        #pragma unroll
        for (int e = 0; e < 16; ++e) {
            // A: lane (m, hi) element e -> row m, K = hi*16 + e
            const int ka = hi * 16 + e;
            a[e]  = sdiff[wv][m][kofs + ka];
            // B: lane (n=m, hi) element e -> column m, K = hi*16 + e; B[k][n] = diff[n][k]
            bm[e] = sdiff[wv][m][kofs + hi * 16 + e];
        }
        acc = __builtin_amdgcn_wmma_f32_16x16x32_f16(false, a, false, bm,
                                                     (short)0, acc, false, false);
    }

    // C/D layout: lane (n, hi), VGPR r -> element (M = r + 8*hi, N = n).
    #pragma unroll
    for (int r = 0; r < 8; ++r)
        sD[wv][r + 8 * hi][m] = acc[r];
    __syncthreads();

    if (lane < 16) {
        const float d2 = sD[wv][lane][lane];
        ew[b * V_ + v0 + lane] = __expf(-d2 * (1.0f / SIGMA_));
    }
}

// ---------------------------------------------------------------------------
// Gather features into BFS order: T[b][v][c] = f[b][c][si[b,v]], T[..][256]=1
// 64 threads/node: 4 strided reads each, one b128 store.
// ---------------------------------------------------------------------------
__global__ void tf_gather_kernel(const float* __restrict__ f,
                                 const int*   __restrict__ si,
                                 float*       __restrict__ T)
{
    const int v = blockIdx.x;
    const int b = blockIdx.y;
    const int t = threadIdx.x;                    // 0..63 -> channels 4t..4t+3
    const int pix = si[b * V_ + v];
    const float* fb = f + (size_t)b * C_ * V_ + pix;
    float4 r;
    r.x = fb[(size_t)(4 * t + 0) * V_];
    r.y = fb[(size_t)(4 * t + 1) * V_];
    r.z = fb[(size_t)(4 * t + 2) * V_];
    r.w = fb[(size_t)(4 * t + 3) * V_];
    float* row = T + (size_t)(b * V_ + v) * CS_;
    reinterpret_cast<float4*>(row)[t] = r;
    if (t == 0) row[C_] = 1.0f;
}

// ---------------------------------------------------------------------------
// Up pass, one level: parent p pulls its <=4 children (4p+1..4p+4).
// Gather form -> deterministic, no atomics. b128 RMW on the parent row.
// ---------------------------------------------------------------------------
__global__ void tf_up_kernel(float* __restrict__ T,
                             const float* __restrict__ ew,
                             int p_begin)
{
    const int p = p_begin + blockIdx.x;
    const int b = blockIdx.y;
    const int c0 = 4 * p + 1;
    if (c0 >= V_) return;

    const float* ewb = ew + b * V_;
    float w[4];
    const float* crow[4];
    #pragma unroll
    for (int k = 0; k < 4; ++k) {
        const int cn = c0 + k;
        const bool ok = cn < V_;
        const int cc = ok ? cn : (V_ - 1);        // safe address, zero weight
        w[k] = ok ? ewb[cn] : 0.0f;
        crow[k] = T + (size_t)(b * V_ + cc) * CS_;
    }
    float* prow = T + (size_t)(b * V_ + p) * CS_;
    const int t = threadIdx.x;                    // 0..63 -> channels 4t..4t+3
    float4 acc = reinterpret_cast<float4*>(prow)[t];
    #pragma unroll
    for (int k = 0; k < 4; ++k) {
        const float4 cv = reinterpret_cast<const float4*>(crow[k])[t];
        acc.x = fmaf(w[k], cv.x, acc.x);
        acc.y = fmaf(w[k], cv.y, acc.y);
        acc.z = fmaf(w[k], cv.z, acc.z);
        acc.w = fmaf(w[k], cv.w, acc.w);
    }
    reinterpret_cast<float4*>(prow)[t] = acc;
    if (t == 0) {                                 // norm channel (c = 256)
        float a = prow[C_];
        #pragma unroll
        for (int k = 0; k < 4; ++k) a = fmaf(w[k], crow[k][C_], a);
        prow[C_] = a;
    }
}

// ---------------------------------------------------------------------------
// Down pass + fused emit, one level (in place): T[v] still holds up[v];
// T[parent] is final. new = up + w*(T[parent] - w*up); out = new / new_norm.
// Leaf level (write_T == 0) skips the T write-back entirely.
// ---------------------------------------------------------------------------
__global__ void tf_down_emit_kernel(float* __restrict__ T,
                                    const float* __restrict__ ew,
                                    const int*   __restrict__ si,
                                    float*       __restrict__ out,
                                    int v_begin, int write_T)
{
    const int v = v_begin + blockIdx.x;
    const int b = blockIdx.y;
    const float w = ew[b * V_ + v];
    const int p = (v - 1) >> 2;
    float*       vrow = T + (size_t)(b * V_ + v) * CS_;
    const float* prow = T + (size_t)(b * V_ + p) * CS_;
    const int t = threadIdx.x;                    // 0..63 -> channels 4t..4t+3

    // norm channel (broadcast loads; every thread needs 1/norm)
    const float un = vrow[C_];
    const float pn = prow[C_];
    const float nn = fmaf(w, pn - w * un, un);
    const float inv = 1.0f / nn;

    const float4 u  = reinterpret_cast<const float4*>(vrow)[t];
    const float4 pf = reinterpret_cast<const float4*>(prow)[t];
    float4 nw;
    nw.x = fmaf(w, pf.x - w * u.x, u.x);
    nw.y = fmaf(w, pf.y - w * u.y, u.y);
    nw.z = fmaf(w, pf.z - w * u.z, u.z);
    nw.w = fmaf(w, pf.w - w * u.w, u.w);

    if (write_T) {
        reinterpret_cast<float4*>(vrow)[t] = nw;
        if (t == 0) vrow[C_] = nn;
    }

    const int pix = si[b * V_ + v];
    float* ob = out + (size_t)b * C_ * V_ + pix;
    ob[(size_t)(4 * t + 0) * V_] = nw.x * inv;
    ob[(size_t)(4 * t + 1) * V_] = nw.y * inv;
    ob[(size_t)(4 * t + 2) * V_] = nw.z * inv;
    ob[(size_t)(4 * t + 3) * V_] = nw.w * inv;
}

// ---------------------------------------------------------------------------
// Root (depth 0) is never touched by the down pass: emit up[0]/norm directly.
// ---------------------------------------------------------------------------
__global__ void tf_root_emit_kernel(const float* __restrict__ T,
                                    const int*   __restrict__ si,
                                    float*       __restrict__ out)
{
    const int b = blockIdx.x;
    const int t = threadIdx.x;                    // 0..63
    const float* row = T + (size_t)b * V_ * CS_;  // v = 0
    const float inv = 1.0f / row[C_];
    const float4 u = reinterpret_cast<const float4*>(row)[t];
    const int pix = si[b * V_];
    float* ob = out + (size_t)b * C_ * V_ + pix;
    ob[(size_t)(4 * t + 0) * V_] = u.x * inv;
    ob[(size_t)(4 * t + 1) * V_] = u.y * inv;
    ob[(size_t)(4 * t + 2) * V_] = u.z * inv;
    ob[(size_t)(4 * t + 3) * V_] = u.w * inv;
}

// ---------------------------------------------------------------------------
extern "C" void kernel_launch(void* const* d_in, const int* in_sizes, int n_in,
                              void* d_out, int out_size, void* d_ws, size_t ws_size,
                              hipStream_t stream)
{
    const float* feature = (const float*)d_in[0];
    const float* embed   = (const float*)d_in[1];
    const int*   si      = (const int*)d_in[2];
    const int*   sp      = (const int*)d_in[3];
    (void)in_sizes; (void)n_in; (void)out_size; (void)ws_size;
    // d_in[4] (depth) and d_in[5] (num_levels) are implied by the fixed
    // 4-ary BFS tree: parent(i) = (i-1)/4, level start L(d) = (4^d - 1)/3.

    float* ew  = (float*)d_ws;                    // B*V floats
    float* T   = ew + (size_t)B_ * V_;            // B*V*CS_ floats
    float* out = (float*)d_out;

    // 1) edge weights (WMMA Gram-diagonal)
    tf_ew_wmma_kernel<<<(B_ * V_) / 64, 128, 0, stream>>>(embed, si, sp, ew);

    // 2) gather features into BFS order (+ norm channel)
    tf_gather_kernel<<<dim3(V_, B_), 64, 0, stream>>>(feature, si, T);

    // level boundaries: L[d] = 4*L[d-1] + 1
    int L[9];
    L[0] = 0;
    for (int d = 1; d <= 8; ++d) {
        int x = 4 * L[d - 1] + 1;
        L[d] = x < V_ ? x : V_;
    }

    // 3) up pass: deepest level first; parents of depth-d nodes are depth d-1
    for (int d = NL_; d >= 1; --d) {
        const int pb = L[d - 1], pe = L[d];
        tf_up_kernel<<<dim3(pe - pb, B_), 64, 0, stream>>>(T, ew, pb);
    }

    // 4) root output (depth 0 bypasses the down pass)
    tf_root_emit_kernel<<<B_, 64, 0, stream>>>(T, si, out);

    // 5) down pass root-outward, with output emission fused per level;
    //    the leaf level skips the T write-back.
    for (int d = 1; d <= NL_; ++d) {
        const int vb = L[d], ve = L[d + 1];
        const int write_T = (d < NL_) ? 1 : 0;
        tf_down_emit_kernel<<<dim3(ve - vb, B_), 64, 0, stream>>>(
            T, ew, si, out, vb, write_T);
    }
}
